// GraphMixupFusion_27307402068681
// MI455X (gfx1250) — matrix operand order repaired and checked by reading, not compiled, not run on hardware
//
#include <hip/hip_runtime.h>
#include <hip/hip_bf16.h>

#define HD 128          // hidden dim
#define PN 500          // nodes per graph
#define PPAD 512        // padded node count for attention tiles
#define NHEADS 4
#define DH 32           // head dim

typedef __attribute__((ext_vector_type(16))) _Float16 v16h;
typedef __attribute__((ext_vector_type(8)))  float    v8f;

// Pack 8 floats (two float4) into fragment slots [base..base+7] as f16.
__device__ __forceinline__ void pack8(v16h& a, int base, float4 u, float4 v) {
  a[base + 0] = (_Float16)u.x; a[base + 1] = (_Float16)u.y;
  a[base + 2] = (_Float16)u.z; a[base + 3] = (_Float16)u.w;
  a[base + 4] = (_Float16)v.x; a[base + 5] = (_Float16)v.y;
  a[base + 6] = (_Float16)v.z; a[base + 7] = (_Float16)v.w;
}

// ---------------------------------------------------------------------------
// CDNA5 WMMA fragment layouts (ISA §7.12.2, wave32):
// 16-bit A 16x32: lane L -> M=L%16, group g=L/16; elem e: K = e<8 ? g*8+e
//                 : 16+g*8+(e-8)  (this code instead uses K = g*16+e: one 16-float run -> float4 x4)
// 16-bit B 32x16: lane L -> N=L%16; elem e: K as for A; code uses K = (L/16)*16 + e (16 contiguous)
// f32  C/D 16x16: lane L -> N=L%16; vgpr r: M = r + 8*(L/16)
// ---------------------------------------------------------------------------

// Out[M,Nw] = act(A[M,K] @ W + bias); W supplied transposed+f16: Wt[Nw][K].
// One wave per 16x16 tile. Address-clamped loads (garbage rows/cols only feed
// stores that are guarded), so the inner loop has zero predication.
__global__ __launch_bounds__(256) void gemm_kernel(
    const float* __restrict__ A, const _Float16* __restrict__ Wt,
    const float* __restrict__ bias, float* __restrict__ Out,
    int M, int K, int Nw, int relu) {
  int wave = blockIdx.x * (blockDim.x >> 5) + (threadIdx.x >> 5);
  int ntiles = (Nw + 15) >> 4;
  int mtiles = (M + 15) >> 4;
  if (wave >= mtiles * ntiles) return;       // wave-uniform exit
  int mt = wave / ntiles, nt = wave % ntiles;
  int lane = threadIdx.x & 31;
  int g = lane >> 4, sl = lane & 15;

  int arow = mt * 16 + sl; if (arow >= M) arow = M - 1;
  int bcol = nt * 16 + sl; if (bcol >= Nw) bcol = Nw - 1;
  const float*    ap = A  + (size_t)arow * K + g * 16;
  const _Float16* bp = Wt + (size_t)bcol * K + g * 16;

  v8f c = {};
  for (int kb = 0; kb < K; kb += 32) {
    float4 q0 = *(const float4*)(ap + kb);
    float4 q1 = *(const float4*)(ap + kb + 4);
    float4 q2 = *(const float4*)(ap + kb + 8);
    float4 q3 = *(const float4*)(ap + kb + 12);
    v16h a;
    pack8(a, 0, q0, q1);
    pack8(a, 8, q2, q3);
    v16h b = *(const v16h*)(bp + kb);
    c = __builtin_amdgcn_wmma_f32_16x16x32_f16(false, a, false, b, (short)0, c,
                                               false, false);
  }
  int col = nt * 16 + sl;
  if (col < Nw) {
    float bv = bias ? bias[col] : 0.f;
#pragma unroll
    for (int r = 0; r < 8; r++) {
      int row = mt * 16 + r + 8 * g;
      if (row < M) {
        float o = c[r] + bv;
        if (relu) o = fmaxf(o, 0.f);
        Out[(size_t)row * Nw + col] = o;
      }
    }
  }
}

// Weight transpose + f32->f16: Wt[n*K + k] = W[k*Nw + n]
__global__ void wtrans_kernel(_Float16* __restrict__ Wt,
                              const float* __restrict__ W, int K, int Nw) {
  int t = blockIdx.x * blockDim.x + threadIdx.x;
  if (t >= K * Nw) return;
  int k = t / Nw, n = t % Nw;
  Wt[(size_t)n * K + k] = (_Float16)W[t];
}

// V transpose per graph, zero-padded: Vt[(g*HD+f)*PPAD + p] = V[(g*PN+p)*HD+f]
__global__ void vtrans_kernel(_Float16* __restrict__ Vt,
                              const float* __restrict__ V, int Gn) {
  size_t idx = (size_t)blockIdx.x * blockDim.x + threadIdx.x;
  if (idx >= (size_t)Gn * HD * PPAD) return;
  int p = (int)(idx & (PPAD - 1));
  size_t gf = idx >> 9;
  int f = (int)(gf & (HD - 1));
  int g = (int)(gf >> 7);
  float val = (p < PN) ? V[((size_t)g * PN + p) * HD + f] : 0.f;
  Vt[idx] = (_Float16)val;
}

// ---------------------------------------------------------------------------
// Adjacency-biased MHA. One WG (4 waves) per (g, head, 16-row tile).
// Stage 1: S[16][512] = QKt/sqrt(dh) + adj (Q-frag hoisted; 32 WMMAs striped)
// Stage 2: row softmax (8-lane subgroups)
// Stage 3: O = P @ Vt (K striped over waves; LDS partial reduce)
// ---------------------------------------------------------------------------
__global__ __launch_bounds__(128) void attn_kernel(
    const float* __restrict__ Q, const float* __restrict__ Km,
    const _Float16* __restrict__ Vt, const float* __restrict__ adj,
    float* __restrict__ Out) {
  __shared__ float S[16][PPAD + 4];
  __shared__ float Op[4][16][32];

  int bid = blockIdx.x;
  int mt = bid & 31, head = (bid >> 5) & 3, g = bid >> 7;
  int wave = threadIdx.x >> 5, lane = threadIdx.x & 31;
  int lg = lane >> 4, ln = lane & 15;
  int row0 = mt * 16;
  int gbase = g * PN;
  const float scale = 0.17677669529663687f;  // 1/sqrt(32)

  // ---- hoisted Q fragment (independent of column tile) ----
  v16h aq;
  {
    int row = row0 + ln; if (row >= PN) row = PN - 1;
    const float* qp = Q + (size_t)(gbase + row) * HD + head * DH + lg * 16;
    pack8(aq, 0, *(const float4*)(qp),      *(const float4*)(qp + 4));
    pack8(aq, 8, *(const float4*)(qp + 8), *(const float4*)(qp + 12));
  }

  // ---- stage 1: score strip ----
  for (int jt = wave; jt < PPAD / 16; jt += 4) {
    int col = jt * 16 + ln;
    int colc = (col < PN) ? col : PN - 1;
    const float* kp = Km + (size_t)(gbase + colc) * HD + head * DH + lg * 16;
    v16h b;
    pack8(b, 0, *(const float4*)(kp),     *(const float4*)(kp + 4));
    pack8(b, 8, *(const float4*)(kp + 8), *(const float4*)(kp + 12));
    v8f c = {};
    c = __builtin_amdgcn_wmma_f32_16x16x32_f16(false, aq, false, b, (short)0, c,
                                               false, false);
#pragma unroll
    for (int r = 0; r < 8; r++) {
      int m = r + 8 * lg;
      int row = row0 + m;
      int rowc = (row < PN) ? row : PN - 1;
      float bv = adj[((size_t)g * PN + rowc) * PN + colc];
      float sv = (col >= PN) ? -1e30f : c[r] * scale + bv;
      S[m][col] = sv;
    }
  }
  __syncthreads();

  // ---- stage 2: softmax, 8 threads per row ----
  {
    int row = threadIdx.x >> 3, sub = threadIdx.x & 7;
    float mx = -1e30f;
    for (int cc = sub; cc < PPAD; cc += 8) mx = fmaxf(mx, S[row][cc]);
#pragma unroll
    for (int o = 4; o >= 1; o >>= 1) mx = fmaxf(mx, __shfl_xor(mx, o, 32));
    float sum = 0.f;
    for (int cc = sub; cc < PPAD; cc += 8) {
      float e = __expf(S[row][cc] - mx);
      S[row][cc] = e;  sum += e;
    }
#pragma unroll
    for (int o = 4; o >= 1; o >>= 1) sum += __shfl_xor(sum, o, 32);
    float inv = 1.f / sum;
    for (int cc = sub; cc < PPAD; cc += 8) S[row][cc] *= inv;
  }
  __syncthreads();

  // ---- stage 3: O = P @ V ----
  v8f o0 = {}, o1 = {};
  for (int ks = wave; ks < PPAD / 32; ks += 4) {
    int kb = ks * 32;
    v16h a;
    {
      const float* sp = &S[ln][kb + lg * 16];
      pack8(a, 0, *(const float4*)(sp),      *(const float4*)(sp + 4));
      pack8(a, 8, *(const float4*)(sp + 8), *(const float4*)(sp + 12));
    }
#pragma unroll
    for (int ntile = 0; ntile < 2; ntile++) {
      const _Float16* vp =
          Vt + ((size_t)g * HD + head * DH + ntile * 16 + ln) * PPAD + kb +
          lg * 16;
      v16h b = *(const v16h*)vp;
      if (ntile == 0)
        o0 = __builtin_amdgcn_wmma_f32_16x16x32_f16(false, a, false, b,
                                                    (short)0, o0, false, false);
      else
        o1 = __builtin_amdgcn_wmma_f32_16x16x32_f16(false, a, false, b,
                                                    (short)0, o1, false, false);
    }
  }
#pragma unroll
  for (int r = 0; r < 8; r++) {
    int m = r + 8 * lg;
    Op[wave][m][ln]      = o0[r];
    Op[wave][m][16 + ln] = o1[r];
  }
  __syncthreads();
  for (int idx = threadIdx.x; idx < 16 * 32; idx += 128) {
    int m = idx >> 5, d = idx & 31;
    float v4 = Op[0][m][d] + Op[1][m][d] + Op[2][m][d] + Op[3][m][d];
    int row = row0 + m;
    if (row < PN) Out[(size_t)(gbase + row) * HD + head * DH + d] = v4;
  }
}

// ---------------------------------------------------------------------------
// Row layernorm (H=128): one wave per row; input = x[row] + (res?res[row]:0)
// ---------------------------------------------------------------------------
__global__ __launch_bounds__(256) void layernorm_kernel(
    float* __restrict__ out, const float* __restrict__ x,
    const float* __restrict__ res, const float* __restrict__ gm,
    const float* __restrict__ bt, int M) {
  int row = blockIdx.x * (blockDim.x >> 5) + (threadIdx.x >> 5);
  if (row >= M) return;
  int lane = threadIdx.x & 31;
  const float* xr = x + (size_t)row * HD;
  const float* rr = res ? res + (size_t)row * HD : nullptr;
  float v[4]; float s = 0.f;
#pragma unroll
  for (int i = 0; i < 4; i++) {
    int c = i * 32 + lane;
    v[i] = xr[c] + (rr ? rr[c] : 0.f);
    s += v[i];
  }
#pragma unroll
  for (int o = 16; o >= 1; o >>= 1) s += __shfl_xor(s, o, 32);
  float mean = s * (1.f / HD);
  float var = 0.f;
#pragma unroll
  for (int i = 0; i < 4; i++) { float d = v[i] - mean; var += d * d; }
#pragma unroll
  for (int o = 16; o >= 1; o >>= 1) var += __shfl_xor(var, o, 32);
  float inv = rsqrtf(var * (1.f / HD) + 1e-5f);
#pragma unroll
  for (int i = 0; i < 4; i++) {
    int c = i * 32 + lane;
    out[(size_t)row * HD + c] = (v[i] - mean) * inv * gm[c] + bt[c];
  }
}

// ---------------------------------------------------------------------------
// Graph kernels
// ---------------------------------------------------------------------------
__global__ void fill_kernel(float* __restrict__ p, size_t n, float v) {
  size_t i = (size_t)blockIdx.x * blockDim.x + threadIdx.x;
  if (i < n) p[i] = v;
}
__global__ void deg_scatter_kernel(float* __restrict__ deg,
                                   const int* __restrict__ dst, int E) {
  int t = blockIdx.x * blockDim.x + threadIdx.x;
  if (t < E) atomicAdd(&deg[dst[t]], 1.f);
}
__global__ void rsqrt_kernel(float* __restrict__ p, int n) {
  int i = blockIdx.x * blockDim.x + threadIdx.x;
  if (i < n) p[i] = rsqrtf(p[i]);
}
__global__ void adj_scatter_kernel(float* __restrict__ adj,
                                   const int* __restrict__ src,
                                   const int* __restrict__ dst, int E) {
  int t = blockIdx.x * blockDim.x + threadIdx.x;
  if (t >= E) return;
  int s = src[t], d = dst[t];
  int g = s / PN;
  atomicAdd(&adj[((size_t)g * PN + (s % PN)) * PN + (d % PN)], 1.f);
}
// out[i] = h[i] * dinv[i]^2  (self-loop term; initializes out)
__global__ void selfloop_kernel(float* __restrict__ out,
                                const float* __restrict__ h,
                                const float* __restrict__ dinv, int n) {
  size_t idx = (size_t)blockIdx.x * blockDim.x + threadIdx.x;
  if (idx >= (size_t)n * HD) return;
  int i = (int)(idx >> 7);
  float w = dinv[i];
  out[idx] = h[idx] * w * w;
}
// out[dst] += h[src]*dinv[src]*dinv[dst]; wave per edge, float4 per lane
__global__ void prop_scatter_kernel(float* __restrict__ out,
                                    const float* __restrict__ h,
                                    const int* __restrict__ src,
                                    const int* __restrict__ dst,
                                    const float* __restrict__ dinv, int E) {
  int t = blockIdx.x * blockDim.x + threadIdx.x;
  int e = t >> 5, lane = t & 31;
  if (e >= E) return;
  int s = src[e], d = dst[e];
  float w = dinv[s] * dinv[d];
  const float4* hs = (const float4*)(h + (size_t)s * HD);
  float4 v = hs[lane];
  float* od = out + (size_t)d * HD + lane * 4;
  atomicAdd(od + 0, v.x * w);
  atomicAdd(od + 1, v.y * w);
  atomicAdd(od + 2, v.z * w);
  atomicAdd(od + 3, v.w * w);
}
__global__ void axpby_kernel(float* __restrict__ out, const float* __restrict__ a,
                             const float* __restrict__ b, float fa, float fb,
                             size_t n) {
  size_t i = (size_t)blockIdx.x * blockDim.x + threadIdx.x;
  if (i < n) out[i] = fa * a[i] + fb * b[i];
}
__global__ void pool_kernel(float* __restrict__ pooled,
                            const float* __restrict__ h) {
  int g = blockIdx.x, j = threadIdx.x;  // 128 threads
  float s = 0.f;
  for (int p = 0; p < PN; p++) s += h[((size_t)g * PN + p) * HD + j];
  pooled[(size_t)g * HD + j] = s;
}

// ---------------------------------------------------------------------------
// Host orchestration
// ---------------------------------------------------------------------------
extern "C" void kernel_launch(void* const* d_in, const int* in_sizes, int n_in,
                              void* d_out, int out_size, void* d_ws, size_t ws_size,
                              hipStream_t stream) {
  (void)n_in; (void)out_size; (void)ws_size;
  const float* x = (const float*)d_in[0];
  const int* ei = (const int*)d_in[1];
  int E = in_sizes[1] / 2;
  const int* src = ei;
  const int* dstp = ei + E;
  int N = in_sizes[0] / HD;
  int G = N / PN;
  auto F = [&](int i) { return (const float*)d_in[i]; };

  // ---- workspace bump allocator (fp32 units) ----
  float* w = (float*)d_ws;
  size_t off = 0;
  auto alloc = [&](size_t n) {
    float* p = w + off;
    off += (n + 63) & ~(size_t)63;
    return p;
  };
  size_t NH = (size_t)N * HD;
  float* dinv = alloc(N);
  float* adj  = alloc((size_t)G * PN * PN);
  float* bXPE = alloc(NH);
  float* bACC = alloc(NH);
  float* bT1  = alloc(NH);
  float* bT1b = alloc(NH);
  float* bT2  = alloc(NH);
  float* bT3  = alloc(NH);
  float* bQ   = alloc(NH);
  float* bK   = alloc(NH);
  float* bV   = alloc(NH);
  float* bAO  = alloc(NH);
  float* bX1  = alloc(NH);
  float* bFFN = alloc((size_t)N * 512);
  float* bGCN = alloc(NH);
  float* bTR  = alloc(NH);
  float* bH   = alloc(NH);
  float* pooled = alloc((size_t)G * HD);
  float* z1   = alloc((size_t)G * HD);
  float* z2   = alloc((size_t)G * 64);
  _Float16* wtScr = (_Float16*)alloc(512 * 128 / 2 + 64);   // transposed weight scratch (f16)
  _Float16* vtBuf = (_Float16*)alloc((size_t)G * HD * PPAD / 2 + 64);  // f16 V^T

  auto fill = [&](float* p, size_t n, float v) {
    fill_kernel<<<dim3((unsigned)((n + 255) / 256)), 256, 0, stream>>>(p, n, v);
  };
  auto ew = [&](float* o, const float* a, const float* b, float fa, float fb,
                size_t n) {
    axpby_kernel<<<dim3((unsigned)((n + 255) / 256)), 256, 0, stream>>>(o, a, b,
                                                                        fa, fb, n);
  };
  auto gemm = [&](const float* A, const float* Wm, const float* bias, float* O,
                  int M, int K, int Nw, int relu) {
    int kn = K * Nw;
    wtrans_kernel<<<dim3((unsigned)((kn + 255) / 256)), 256, 0, stream>>>(
        wtScr, Wm, K, Nw);
    int tiles = ((M + 15) / 16) * ((Nw + 15) / 16);
    gemm_kernel<<<dim3((unsigned)((tiles + 7) / 8)), 256, 0, stream>>>(
        A, wtScr, bias, O, M, K, Nw, relu);
  };
  auto prop = [&](const float* hin, float* outb) {
    selfloop_kernel<<<dim3((unsigned)((NH + 255) / 256)), 256, 0, stream>>>(
        outb, hin, dinv, N);
    size_t th = (size_t)E * 32;
    prop_scatter_kernel<<<dim3((unsigned)((th + 255) / 256)), 256, 0, stream>>>(
        outb, hin, src, dstp, dinv, E);
  };
  // multiscale SGConv stack: params at index pb (2 scales x 3 convs x (W,b), fc)
  auto multiscale = [&](const float* hin, int pb, float* outb) {
    fill(bACC, NH, 0.f);
    for (int c = 0; c < 2; c++) {
      int KK = c + 1;
      for (int j = 0; j < 3; j++) {
        const float* cur = (j == 0) ? hin : (const float*)bT2;
        for (int k = 0; k < KK; k++) {
          float* nxt = (cur == (const float*)bT1) ? bT1b : bT1;
          prop(cur, nxt);
          cur = nxt;
        }
        gemm(cur, F(pb + c * 6 + j * 2), F(pb + c * 6 + j * 2 + 1), bT2,
             N, HD, HD, 0);
      }
      ew(bACC, bACC, bT2, 1.f, 1.f, NH);
    }
    ew(bACC, bACC, bACC, 0.25f, 0.25f, NH);  // /= 2
    gemm(bACC, F(pb + 12), F(pb + 13), outb, N, HD, HD, 0);
  };
  // transformer block: params at index tb (q,k,v,o,(ln1),f1,f2,(ln2))
  auto transformer = [&](const float* xpe, int tb, float* outb) {
    gemm(xpe, F(tb + 0), F(tb + 1), bQ, N, HD, HD, 0);
    gemm(xpe, F(tb + 2), F(tb + 3), bK, N, HD, HD, 0);
    gemm(xpe, F(tb + 4), F(tb + 5), bV, N, HD, HD, 0);
    size_t vtN = (size_t)G * HD * PPAD;
    vtrans_kernel<<<dim3((unsigned)((vtN + 255) / 256)), 256, 0, stream>>>(
        vtBuf, bV, G);
    attn_kernel<<<dim3((unsigned)(G * NHEADS * (PPAD / 16))), 128, 0, stream>>>(
        bQ, bK, vtBuf, adj, bAO);
    gemm(bAO, F(tb + 6), F(tb + 7), bT3, N, HD, HD, 0);
    layernorm_kernel<<<dim3((unsigned)((N + 7) / 8)), 256, 0, stream>>>(
        bX1, bT3, xpe, F(tb + 8), F(tb + 9), N);
    gemm(bX1, F(tb + 10), F(tb + 11), bFFN, N, HD, 512, 1);
    gemm(bFFN, F(tb + 12), F(tb + 13), bT3, N, 512, HD, 0);
    layernorm_kernel<<<dim3((unsigned)((N + 7) / 8)), 256, 0, stream>>>(
        outb, bT3, bX1, F(tb + 14), F(tb + 15), N);
  };

  // ---- preprocessing: degrees, dinv, dense adjacency ----
  fill(dinv, N, 1.0f);  // self loop
  deg_scatter_kernel<<<dim3((unsigned)((E + 255) / 256)), 256, 0, stream>>>(
      dinv, dstp, E);
  rsqrt_kernel<<<dim3((unsigned)((N + 255) / 256)), 256, 0, stream>>>(dinv, N);
  hipMemsetAsync(adj, 0, (size_t)G * PN * PN * sizeof(float), stream);
  adj_scatter_kernel<<<dim3((unsigned)((E + 255) / 256)), 256, 0, stream>>>(
      adj, src, dstp, E);

  // ---- top level: GCN branch + transformer branch, alpha=0.5 mixup ----
  gemm(x, F(3), F(4), bXPE, N, HD, HD, 0);     // pe0
  multiscale(x, 21, bGCN);                     // gcn1
  transformer(bXPE, 5, bTR);                   // trans1
  ew(bH, bGCN, bTR, 0.5f, 0.5f, NH);

  // ---- SelfMixupFusionLayer enc0, alpha1=0.4 ----
  multiscale(bH, 37, bGCN);                    // enc0.gcn
  gemm(bH, F(35), F(36), bXPE, N, HD, HD, 0);  // enc0.pe
  transformer(bXPE, 51, bTR);                  // enc0.trans
  ew(bH, bGCN, bTR, 0.4f, 0.6f, NH);

  // ---- global_add_pool + MLP head ----
  pool_kernel<<<dim3((unsigned)G), 128, 0, stream>>>(pooled, bH);
  gemm(pooled, F(67), F(68), z1, G, HD, HD, 1);
  gemm(z1, F(69), F(70), z2, G, HD, 64, 1);
  gemm(z2, F(71), F(72), (float*)d_out, G, 64, 10, 0);
}
